// Generator_63333587746891
// MI455X (gfx1250) — compile-verified, matrix operand order checked by reading, not run on hardware
//
#include <hip/hip_runtime.h>
#include <math.h>
#include <stdint.h>

// Problem constants (match reference)
#define NN   4096
#define EE   16384
#define LR   64
#define HR   128
#define EPSB 1e-3f

typedef __attribute__((ext_vector_type(16))) _Float16 v16h;
typedef __attribute__((ext_vector_type(8)))  float    v8f;

typedef __attribute__((ext_vector_type(4))) unsigned int tdm_u32x4;
typedef __attribute__((ext_vector_type(8))) int          tdm_i32x8;
typedef __attribute__((ext_vector_type(4))) int          tdm_i32x4;

// 32-bit LDS-address-space pointer type: loads through this can only lower
// to DS instructions (not FLAT).
typedef __attribute__((address_space(3))) _Float16 lds_f16;

// ---------------------------------------------------------------------------
// Workspace layout (float elements)
//   agg1 [N*LR] | cnt [N] | agg3 [N*HR]   <- zeroed each call
//   x1   [N*LR] f32 | x3h [N*HR] f16
// ---------------------------------------------------------------------------
#define OFF_AGG1 0
#define OFF_CNT  (OFF_AGG1 + NN * LR)            // 262144
#define OFF_AGG3 (OFF_CNT + NN)                  // 266240
#define OFF_X1   (OFF_AGG3 + NN * HR)            // 790528
#define OFF_X3H  (OFF_X1 + NN * LR)              // 1052672 (f16 region)
#define ZERO_CNT OFF_X1                          // contiguous zero region

// ---------------------------------------------------------------------------
// WMMA fragment index helper (ISA 7.12.2, 16-bit A 16x32 / B 32x16):
// lane holds M (or N) = lane&15; lane[4] selects the 8-K half of each 16-K
// group; v16h element e -> VGPR e/2, K = 16*(v/4) + 8*lane[4] + 2*(v&3) + e&1.
// ---------------------------------------------------------------------------
__device__ inline int frag_k(int e, int lane) {
    int v = e >> 1;
    return ((v >= 4) ? 16 : 0) + ((lane & 16) ? 8 : 0) + 2 * (v & 3) + (e & 1);
}

__device__ inline v16h load_frag_a(const float* __restrict__ A, int lda,
                                   int row0, int k0, int lane) {
    v16h f;
    int m = row0 + (lane & 15);
#pragma unroll
    for (int e = 0; e < 16; ++e)
        f[e] = (_Float16)A[m * lda + k0 + frag_k(e, lane)];
    return f;
}

__device__ inline v16h load_frag_b(const float* __restrict__ B, int ldb,
                                   int k0, int col0, int lane) {
    v16h f;
    int n = col0 + (lane & 15);
#pragma unroll
    for (int e = 0; e < 16; ++e)
        f[e] = (_Float16)B[(k0 + frag_k(e, lane)) * ldb + n];
    return f;
}

__device__ inline float sigmoidf_fast(float v) {
    return 1.0f / (1.0f + __expf(-v));
}

// ---------------------------------------------------------------------------
// K0: zero a float region
// ---------------------------------------------------------------------------
__global__ void k_zero(float* __restrict__ p, int count) {
    int i = blockIdx.x * blockDim.x + threadIdx.x;
    if (i < count) p[i] = 0.0f;
}

// ---------------------------------------------------------------------------
// K1/K3: per-edge NNConv message with fused ReLU weight generation.
// One wave32 per edge; each lane owns OUT/32 output columns; fp32 atomics
// implement the scatter part of scatter-mean.  Never materializes the
// E x IN x OUT per-edge weight tensor (would be 768MB of traffic).
// ---------------------------------------------------------------------------
template <int IN, int OUT>
__global__ void k_edge_msg(const float* __restrict__ xin,
                           const int*   __restrict__ edge_index,
                           const float* __restrict__ edge_attr,
                           const float* __restrict__ w_nn,
                           const float* __restrict__ b_nn,
                           float* __restrict__ agg,
                           float* __restrict__ cnt,   // nullptr on layer 2
                           int n_edges) {
    int gtid = blockIdx.x * blockDim.x + threadIdx.x;
    int edge = gtid >> 5;
    int lane = threadIdx.x & 31;
    if (edge >= n_edges) return;

    int   src = edge_index[edge];
    int   dst = edge_index[n_edges + edge];
    float a   = edge_attr[edge];

    constexpr int PER = OUT / 32;   // outputs per lane (2 or 4)
    float acc[PER];
#pragma unroll
    for (int p = 0; p < PER; ++p) acc[p] = 0.0f;

    const float* xrow = xin + src * IN;
#pragma unroll 4
    for (int i = 0; i < IN; ++i) {
        float xi = xrow[i];
        const float* wr = w_nn + i * OUT;
        const float* br = b_nn + i * OUT;
#pragma unroll
        for (int p = 0; p < PER; ++p) {
            int   o = lane + 32 * p;
            float w = fmaf(a, wr[o], br[o]);   // edge MLP: a*w + b
            w = w > 0.0f ? w : 0.0f;           // ReLU
            acc[p] = fmaf(xi, w, acc[p]);
        }
    }
#pragma unroll
    for (int p = 0; p < PER; ++p)
        atomicAdd(&agg[dst * OUT + lane + 32 * p], acc[p]);
    if (cnt != nullptr && lane == 0)
        atomicAdd(&cnt[dst], 1.0f);
}

// ---------------------------------------------------------------------------
// K2/K4: node update.  h = agg/max(cnt,1) + x@root + bias; BN eval; sigmoid.
// One wave per 16-row block; x@root via v_wmma_f32_16x16x32_f16 (K=64 -> 2
// WMMA K-steps per 16-wide C tile; OUT/16 tiles along N).  OutT=f32 for x1
// (consumed by the VALU edge kernel), OutT=f16 for x3 (consumed only by the
// f16 WMMA Gram kernel).
// ---------------------------------------------------------------------------
template <int OUT, typename OutT>
__global__ void k_node_update(const float* __restrict__ xin,   // [N, LR]
                              const float* __restrict__ root,  // [LR, OUT]
                              const float* __restrict__ agg,   // [N, OUT]
                              const float* __restrict__ cnt,   // [N]
                              const float* __restrict__ bias,
                              const float* __restrict__ gamma,
                              const float* __restrict__ beta,
                              const float* __restrict__ mean,
                              const float* __restrict__ var,
                              OutT* __restrict__ xout) {       // [N, OUT]
    constexpr int NT = OUT / 16;       // C tiles along N (4 or 8)
    int lane = threadIdx.x;            // blockDim.x == 32, no divergence
    int row0 = blockIdx.x * 16;

    v8f c[NT];
#pragma unroll
    for (int t = 0; t < NT; ++t) c[t] = v8f{};

#pragma unroll
    for (int kk = 0; kk < LR; kk += 32) {
        v16h af = load_frag_a(xin, LR, row0, kk, lane);
#pragma unroll
        for (int t = 0; t < NT; ++t) {
            v16h bf = load_frag_b(root, OUT, kk, t * 16, lane);
            c[t] = __builtin_amdgcn_wmma_f32_16x16x32_f16(
                false, af, false, bf, (short)0, c[t], false, false);
        }
    }

    // C layout: VGPR r -> M = r + 8*lane[4]; N = lane&15
    int mbase = (lane & 16) ? 8 : 0;
    int nlane = lane & 15;
#pragma unroll
    for (int t = 0; t < NT; ++t) {
#pragma unroll
        for (int r = 0; r < 8; ++r) {
            int   row = row0 + mbase + r;
            int   col = t * 16 + nlane;
            float deg = cnt[row];
            deg = deg > 1.0f ? deg : 1.0f;
            float h  = c[t][r] + agg[row * OUT + col] / deg + bias[col];
            float hn = gamma[col] * (h - mean[col]) * rsqrtf(var[col] + EPSB)
                     + beta[col];
            xout[row * OUT + col] = (OutT)sigmoidf_fast(hn);
        }
    }
}

// ---------------------------------------------------------------------------
// K5: out = x3^T @ x3  [HR x HR], K = NN, K-split across KSPLIT blocks.
// Each block: 8 waves; K-chunks of 64 rows of x3h (f16, 16KB) staged into
// LDS by the Tensor Data Mover (one D# per wave covering 8 rows), waves wait
// on TENSORcnt, barrier, then each wave computes its 16-row M-strip of the
// 128x128 output (8 WMMA tiles) from LDS.  Partial C tiles are combined with
// fp32 global atomics into the pre-zeroed output.
//
// The TDM builtin carries the LDS destination only inside the D# descriptor,
// so the compiler sees no def of `tile`.  We (a) put the real LDS offset of
// `tile` in the descriptor (generic-address bits [31:0] are the LDS offset
// for the shared aperture, ISA 10.2) and (b) escape an addrspace(3) pointer
// to the tile through an asm memory clobber after s_wait_tensorcnt, then read
// fragments through that pointer: AS(3) loads can only lower to ds_load_*,
// keeping the traffic on the LDS pipe (last round's generic-pointer escape
// degraded them to flat_load_u16).
// ---------------------------------------------------------------------------
#define KSPLIT 16
#define KCHUNK 64

__global__ void k_out_gemm(const _Float16* __restrict__ x3h,
                           float* __restrict__ out) {
    __shared__ _Float16 tile[KCHUNK * HR];   // 16 KB

    int lane = threadIdx.x & 31;
    int wave = threadIdx.x >> 5;             // 0..7 -> M strip
    int kbase = blockIdx.x * (NN / KSPLIT);  // 256 K-rows per block

    v8f c[HR / 16];
#pragma unroll
    for (int t = 0; t < HR / 16; ++t) c[t] = v8f{};

    for (int kc = 0; kc < NN / KSPLIT; kc += KCHUNK) {
        int k0 = kbase + kc;

        // LDS byte offset of the tile (generic addr bits [31:0] in the
        // shared aperture) -> 32-bit addrspace(3) pointer.
        unsigned  lds_base = (unsigned)(uintptr_t)(void*)tile;
        lds_f16*  tl       = (lds_f16*)lds_base;

        // ---- stage x3h[k0 .. k0+KCHUNK) x [0..HR) into LDS ----
#if defined(__has_builtin) && __has_builtin(__builtin_amdgcn_tensor_load_to_lds)
        {
            // Each wave DMAs its own 8-row slice (TENSORcnt is per-wave;
            // TDM ignores EXEC so every wave issues exactly one descriptor).
            const int rpw = KCHUNK / 8;                       // rows per wave
            unsigned lds_off = lds_base +
                (unsigned)(wave * rpw * HR * (int)sizeof(_Float16));
            unsigned long long ga =
                (unsigned long long)(uintptr_t)(x3h + (size_t)(k0 + wave * rpw) * HR);

            // D# group 0: count=1 | lds_addr | global_addr[56:0] | type=2
            tdm_u32x4 g0 = { 1u,
                             lds_off,
                             (unsigned)ga,
                             (unsigned)((ga >> 32) & 0x01FFFFFFull) | (2u << 30) };
            // D# group 1 (ISA 8.4): data_size=1 (2B); tensor_dim0=HR;
            // tensor_dim1=NN; tile_dim0=HR; tile_dim1=rpw; dim0_stride=HR.
            tdm_i32x8 g1 = { (int)(1u << 16),          // data_size=2B
                             (int)((unsigned)HR << 16),// tensor_dim0[15:0]
                             (int)((unsigned)NN << 16),// tensor_dim1[15:0]
                             (int)((unsigned)HR << 16),// tile_dim0
                             rpw,                      // tile_dim1
                             HR,                       // tensor_dim0_stride lo
                             0, 0 };
            tdm_i32x4 gz = { 0, 0, 0, 0 };             // 2-D tensor: groups 2/3 unused
#if __has_include(<hip/amd_detail/amd_gfx1250_TDM.h>)
            tdm_i32x8 gz8 = { 0, 0, 0, 0, 0, 0, 0, 0 };
            __builtin_amdgcn_tensor_load_to_lds(g0, g1, gz, gz, gz8, 0);
#else
            __builtin_amdgcn_tensor_load_to_lds(g0, g1, gz, gz, 0);
#endif
            __builtin_amdgcn_s_wait_tensorcnt(0);
        }
        // The TDM engine has written the tile; escape the LDS pointer so the
        // ds_load fragment reads below cannot be folded away or hoisted.
        asm volatile("" : "+r"(tl) : : "memory");
#else
        for (int i = threadIdx.x; i < KCHUNK * HR; i += (int)blockDim.x)
            tile[i] = x3h[(size_t)k0 * HR + i];
#endif
        __syncthreads();

        // ---- WMMA over the staged chunk (fragments fed from the LDS pipe) --
#pragma unroll
        for (int ks = 0; ks < KCHUNK; ks += 32) {
            v16h af;   // A[m][k] = x3[k][wave*16 + m]  (transposed view)
#pragma unroll
            for (int e = 0; e < 16; ++e)
                af[e] = tl[(ks + frag_k(e, lane)) * HR + wave * 16 + (lane & 15)];
#pragma unroll
            for (int t = 0; t < HR / 16; ++t) {
                v16h bf;   // B[k][n] = x3[k][t*16 + n]
#pragma unroll
                for (int e = 0; e < 16; ++e)
                    bf[e] = tl[(ks + frag_k(e, lane)) * HR + t * 16 + (lane & 15)];
                c[t] = __builtin_amdgcn_wmma_f32_16x16x32_f16(
                    false, af, false, bf, (short)0, c[t], false, false);
            }
        }
        __syncthreads();   // before next chunk overwrites the tile
    }

    // ---- combine K-split partials ----
    int mbase = (lane & 16) ? 8 : 0;
    int nlane = lane & 15;
#pragma unroll
    for (int t = 0; t < HR / 16; ++t)
#pragma unroll
        for (int r = 0; r < 8; ++r)
            atomicAdd(&out[(wave * 16 + mbase + r) * HR + t * 16 + nlane],
                      c[t][r]);
}

// ---------------------------------------------------------------------------
extern "C" void kernel_launch(void* const* d_in, const int* in_sizes, int n_in,
                              void* d_out, int out_size, void* d_ws, size_t ws_size,
                              hipStream_t stream) {
    const float* x          = (const float*)d_in[0];
    const int*   edge_index = (const int*)  d_in[1];
    const float* edge_attr  = (const float*)d_in[2];
    const float* w_nn1      = (const float*)d_in[3];
    const float* b_nn1      = (const float*)d_in[4];
    const float* root1      = (const float*)d_in[5];
    const float* bias1      = (const float*)d_in[6];
    const float* bn1_gamma  = (const float*)d_in[7];
    const float* bn1_beta   = (const float*)d_in[8];
    const float* bn1_mean   = (const float*)d_in[9];
    const float* bn1_var    = (const float*)d_in[10];
    const float* w_nn3      = (const float*)d_in[11];
    const float* b_nn3      = (const float*)d_in[12];
    const float* root3      = (const float*)d_in[13];
    const float* bias3      = (const float*)d_in[14];
    const float* bn3_gamma  = (const float*)d_in[15];
    const float* bn3_beta   = (const float*)d_in[16];
    const float* bn3_mean   = (const float*)d_in[17];
    const float* bn3_var    = (const float*)d_in[18];

    float*     ws   = (float*)d_ws;
    float*     agg1 = ws + OFF_AGG1;
    float*     cnt  = ws + OFF_CNT;
    float*     agg3 = ws + OFF_AGG3;
    float*     x1   = ws + OFF_X1;
    _Float16*  x3h  = (_Float16*)(ws + OFF_X3H);
    float*     out  = (float*)d_out;

    // K0: zero agg1 | cnt | agg3 (contiguous) and the K-split output
    k_zero<<<(ZERO_CNT + 255) / 256, 256, 0, stream>>>(ws, ZERO_CNT);
    k_zero<<<(HR * HR + 255) / 256, 256, 0, stream>>>(out, HR * HR);

    // K1: layer-1 edge messages (one wave per edge)
    k_edge_msg<LR, LR><<<(EE * 32 + 255) / 256, 256, 0, stream>>>(
        x, edge_index, edge_attr, w_nn1, b_nn1, agg1, cnt, EE);

    // K2: layer-1 node update (WMMA x@root1 + mean/bias/BN/sigmoid) -> f32 x1
    k_node_update<LR, float><<<NN / 16, 32, 0, stream>>>(
        x, root1, agg1, cnt, bias1, bn1_gamma, bn1_beta, bn1_mean, bn1_var, x1);

    // K3: layer-2 edge messages
    k_edge_msg<LR, HR><<<(EE * 32 + 255) / 256, 256, 0, stream>>>(
        x1, edge_index, edge_attr, w_nn3, b_nn3, agg3, nullptr, EE);

    // K4: layer-2 node update (WMMA x1@root3) -> f16 x3
    k_node_update<HR, _Float16><<<NN / 16, 32, 0, stream>>>(
        x1, root3, agg3, cnt, bias3, bn3_gamma, bn3_beta, bn3_mean, bn3_var, x3h);

    // K5: Gram matrix x3^T @ x3 (TDM-staged LDS tiles + WMMA + K-split atomics)
    k_out_gemm<<<KSPLIT, 256, 0, stream>>>(x3h, out);
}
